// MoE_90297392431448
// MI455X (gfx1250) — matrix operand order repaired and checked by reading, not compiled, not run on hardware
//
#include <hip/hip_runtime.h>
#include <hip/hip_bf16.h>
#include <math.h>

// ---------------------------------------------------------------------------
// MoE forward for gfx1250 (MI455X): bf16 WMMA (v_wmma_f32_16x16x32_bf16),
// f32 accumulate, TDM tensor_load_to_lds for the x tile (TENSORcnt),
// global_load_async_to_lds_b128 for the b2 bias table (ASYNCcnt, wait
// deferred to the epilogue so the copy hides behind the WMMA main loop).
// ---------------------------------------------------------------------------

typedef __attribute__((ext_vector_type(16))) __bf16   v16bf;
typedef __attribute__((ext_vector_type(8)))  __bf16   v8bf;
typedef __attribute__((ext_vector_type(8)))  float    v8f;
typedef __attribute__((ext_vector_type(4)))  unsigned u32x4;
typedef __attribute__((ext_vector_type(8)))  unsigned u32x8;

constexpr int Dd = 1024;   // model dim
constexpr int Hh = 4096;   // hidden dim
constexpr int Ee = 8;      // experts
constexpr int TM = 16;     // token tile (M)
constexpr int HC = 128;    // H chunk per stage = 8 waves * 16

union V16U { v16bf v; v8bf h[2]; };

// Build a 16x32 bf16 A-fragment for one lane from row-major (K-contiguous)
// storage. ISA layout: lane m = L%16; VGPR0-3 hold K = k0+8*(L/16)..+7,
// VGPR4-7 hold K = k0+8*(L/16)+16..+23  -> two contiguous 16B chunks.
__device__ __forceinline__ v16bf ldA(const __bf16* rowBase, int k0) {
  V16U u;
  u.h[0] = *(const v8bf*)(rowBase + k0);
  u.h[1] = *(const v8bf*)(rowBase + k0 + 16);
  return u.v;
}

// ---------------------------------------------------------------------------
// 1) x (f32) -> bf16
// ---------------------------------------------------------------------------
__global__ __launch_bounds__(256) void cvt_x_kernel(const float* __restrict__ in,
                                                    __bf16* __restrict__ out, int n) {
  for (int i = blockIdx.x * blockDim.x + threadIdx.x; i < n;
       i += gridDim.x * blockDim.x)
    out[i] = (__bf16)in[i];
}

// ---------------------------------------------------------------------------
// 2) Per-expert transpose + convert:  in [E][R][C] f32  ->  out [E][C][R] bf16
//    (gives the WMMA B operand contiguous-K rows)
// ---------------------------------------------------------------------------
__global__ __launch_bounds__(256) void transpose_cvt_kernel(
    const float* __restrict__ in, __bf16* __restrict__ out, int R, int C) {
  __shared__ float tile[32][33];
  const int e  = blockIdx.z;
  const int c0 = blockIdx.x * 32;
  const int r0 = blockIdx.y * 32;
  const float* ip = in  + (size_t)e * R * C;
  __bf16*      op = out + (size_t)e * R * C;
  const int tx = threadIdx.x, ty = threadIdx.y;
  #pragma unroll
  for (int i = ty; i < 32; i += 8)
    tile[i][tx] = ip[(size_t)(r0 + i) * C + (c0 + tx)];
  __syncthreads();
  #pragma unroll
  for (int i = ty; i < 32; i += 8)
    op[(size_t)(c0 + i) * R + (r0 + tx)] = (__bf16)tile[tx][i];
}

// ---------------------------------------------------------------------------
// 3) Gating: one wave32 per token. softmax over E=8, top-2, L1 renorm.
// ---------------------------------------------------------------------------
__global__ __launch_bounds__(256) void gate_kernel(
    const float* __restrict__ x, const float* __restrict__ Wg,
    const float* __restrict__ bg, float* __restrict__ g, int Ntok) {
  const int wavesPerBlk = blockDim.x >> 5;
  const int token = blockIdx.x * wavesPerBlk + (threadIdx.x >> 5);
  const int lane  = threadIdx.x & 31;
  if (token >= Ntok) return;
  const float* xp = x + (size_t)token * Dd;

  float acc[Ee];
  #pragma unroll
  for (int e = 0; e < Ee; ++e) acc[e] = 0.f;
  for (int d = lane; d < Dd; d += 32) {
    float xv = xp[d];
    #pragma unroll
    for (int e = 0; e < Ee; ++e) acc[e] = fmaf(xv, Wg[d * Ee + e], acc[e]);
  }
  #pragma unroll
  for (int off = 16; off > 0; off >>= 1) {
    #pragma unroll
    for (int e = 0; e < Ee; ++e) acc[e] += __shfl_xor(acc[e], off, 32);
  }
  if (lane == 0) {
    float p[Ee];
    float mx = acc[0] + bg[0];
    #pragma unroll
    for (int e = 0; e < Ee; ++e) { p[e] = acc[e] + bg[e]; mx = fmaxf(mx, p[e]); }
    float s = 0.f;
    #pragma unroll
    for (int e = 0; e < Ee; ++e) { p[e] = expf(p[e] - mx); s += p[e]; }
    float inv = 1.f / s;
    #pragma unroll
    for (int e = 0; e < Ee; ++e) p[e] *= inv;
    int i1 = 0;
    #pragma unroll
    for (int e = 1; e < Ee; ++e) if (p[e] > p[i1]) i1 = e;
    int i2 = (i1 == 0) ? 1 : 0;
    #pragma unroll
    for (int e = 0; e < Ee; ++e) if (e != i1 && p[e] > p[i2]) i2 = e;
    float denom = fmaxf(p[i1] + p[i2], 1e-12f);
    #pragma unroll
    for (int e = 0; e < Ee; ++e)
      g[(size_t)token * Ee + e] = (e == i1 || e == i2) ? p[e] / denom : 0.f;
  }
}

// ---------------------------------------------------------------------------
// 4) Fused dual-GEMM FFN with WMMA. 8 waves / block, 16-token tile.
//    xb   [Ntok][D]  bf16   (staged to LDS by the Tensor Data Mover)
//    W1T  [E][H][D]  bf16   (B operand of GEMM1, K=D contiguous)
//    W2T  [E][D][H]  bf16   (B operand of GEMM2, K=H contiguous)
// ---------------------------------------------------------------------------
__global__ __launch_bounds__(256) void moe_ffn_kernel(
    const __bf16* __restrict__ xb, const __bf16* __restrict__ W1T,
    const __bf16* __restrict__ W2T, const float* __restrict__ b1,
    const float* __restrict__ b2, const float* __restrict__ gbuf,
    float* __restrict__ out) {
  __shared__ alignas(16) __bf16 xs[TM][Dd];     // 32 KB: token tile (TDM dest)
  __shared__ alignas(16) __bf16 hbuf[TM][HC];   // 4 KB: gelu(h)*g chunk
  __shared__ alignas(16) float  b2s[Ee * Dd];   // 32 KB: b2 (async dest)
  __shared__ float  gs[TM][Ee];                 // gate weights for this tile

  const int tid    = threadIdx.x;
  const int wave   = tid >> 5;
  const int lane   = tid & 31;
  const int lm     = lane & 15;   // lane's M (A) or N (B/C) index
  const int lh     = lane >> 4;   // lane half -> K-offset group
  const int token0 = blockIdx.x * TM;

  // --- TDM: one descriptor moves the whole 16x1024 bf16 x tile to LDS ---
  if (wave == 0) {
    unsigned       ldsoff = (unsigned)(size_t)(&xs[0][0]);
    unsigned long long ga = (unsigned long long)(xb + (size_t)token0 * Dd);
    u32x4 g0;
    g0[0] = 1u;                                    // count=1, user descriptor
    g0[1] = ldsoff;                                // lds_addr (bytes)
    g0[2] = (unsigned)ga;                          // global_addr[31:0]
    g0[3] = (unsigned)(ga >> 32) | (2u << 30);     // addr[56:32] | type=2
    u32x8 g1;
    g1[0] = 1u << 16;                              // data_size=1 (2 bytes)
    g1[1] = ((unsigned)Dd & 0xFFFFu) << 16;        // tensor_dim0[15:0]
    g1[2] = ((unsigned)Dd >> 16) | ((unsigned)TM << 16);  // dim0 hi | dim1 lo
    g1[3] = ((unsigned)TM >> 16) | ((unsigned)Dd << 16);  // dim1 hi | tile_dim0
    g1[4] = (unsigned)TM;                          // tile_dim1 | tile_dim2=0
    g1[5] = (unsigned)Dd;                          // tensor_dim0_stride lo32
    g1[6] = 0u;                                    // stride0 hi | stride1 lo
    g1[7] = 0u;                                    // stride1 hi
    asm volatile("tensor_load_to_lds %0, %1" :: "s"(g0), "s"(g1) : "memory");
  }

  // --- ASYNC: stage b2 [E][D] f32 (32 KB) into LDS; wait deferred to end ---
  {
    unsigned           ldsb = (unsigned)(size_t)(&b2s[0]);
    unsigned long long gb   = (unsigned long long)b2;
    #pragma unroll
    for (int i = 0; i < (Ee * Dd * 4) / (256 * 16); ++i) {   // 8 x 16B/thread
      unsigned boff = (unsigned)(tid + i * 256) * 16u;
      asm volatile("global_load_async_to_lds_b128 %0, %1, off"
                   :: "v"(ldsb + boff), "v"(gb + boff) : "memory");
    }
  }

  if (tid < TM * Ee) ((float*)gs)[tid] = gbuf[(size_t)token0 * Ee + tid];
  if (wave == 0) __builtin_amdgcn_s_wait_tensorcnt(0);   // x tile landed
  __syncthreads();

  const v8f vzero = {0.f, 0.f, 0.f, 0.f, 0.f, 0.f, 0.f, 0.f};
  const int dcol0 = wave * 128;   // this wave's 128-wide slice of D
  v8f acc[8];
  #pragma unroll
  for (int n = 0; n < 8; ++n) acc[n] = vzero;   // accumulates ALL experts

  for (int e = 0; e < Ee; ++e) {
    const __bf16* w1e = W1T + (size_t)e * Hh * Dd;
    const __bf16* w2e = W2T + (size_t)e * Dd * Hh;
    for (int hc = 0; hc < Hh / HC; ++hc) {
      // ---- GEMM1: this wave computes h-tile [16 x 16] at h0, K = D ----
      const int h0 = hc * HC + wave * 16;
      const int nh = h0 + lm;                              // lane's h column
      const __bf16* brow = w1e + (size_t)nh * Dd + lh * 8;
      const __bf16* arow = &xs[lm][lh * 8];
      v8f hacc0 = vzero, hacc1 = vzero;                    // 2 chains for ILP
      #pragma unroll 4
      for (int k0 = 0; k0 < Dd; k0 += 64) {
        v16bf a0 = ldA(arow, k0);
        v16bf bb0 = ldA(brow, k0);
        hacc0 = __builtin_amdgcn_wmma_f32_16x16x32_bf16(
            false, a0, false, bb0, (short)0, hacc0, false, false);
        v16bf a1 = ldA(arow, k0 + 32);
        v16bf bb1 = ldA(brow, k0 + 32);
        hacc1 = __builtin_amdgcn_wmma_f32_16x16x32_bf16(
            false, a1, false, bb1, (short)0, hacc1, false, false);
      }
      v8f hacc = hacc0 + hacc1;
      const float bias1 = b1[e * Hh + nh];   // uniform per lane (fixed n)

      __syncthreads();   // previous chunk's hbuf fully consumed
      #pragma unroll
      for (int r = 0; r < 8; ++r) {
        const int m = r + 8 * lh;            // C layout: M = r + 8*(lane/16)
        float v = hacc[r] + bias1;
        float gel = 0.5f * v * (1.0f + erff(v * 0.70710678118654752f));
        hbuf[m][wave * 16 + lm] = (__bf16)(gel * gs[m][e]);  // gate folded in
      }
      __syncthreads();   // hbuf ready

      // ---- GEMM2: acc[n] += hbuf[16 x 128] @ W2T-slice, K = 128 ----
      V16U af[4];
      #pragma unroll
      for (int kk = 0; kk < 4; ++kk) {
        af[kk].h[0] = *(const v8bf*)&hbuf[lm][kk * 32 + lh * 8];
        af[kk].h[1] = *(const v8bf*)&hbuf[lm][kk * 32 + lh * 8 + 16];
      }
      const __bf16* w2base =
          w2e + (size_t)(dcol0 + lm) * Hh + hc * HC + lh * 8;
      #pragma unroll
      for (int kk = 0; kk < 4; ++kk) {       // kk outer, n inner -> WMMA ILP=8
        #pragma unroll
        for (int n = 0; n < 8; ++n) {
          v16bf b = ldA(w2base + (size_t)(n * 16) * Hh, kk * 32);
          acc[n] = __builtin_amdgcn_wmma_f32_16x16x32_bf16(
              false, af[kk].v, false, b, (short)0, acc[n], false, false);
        }
      }
      __builtin_prefetch(w2base + HC, 0, 1);  // next H-chunk of W2
    }
  }

  // b2 async copy must be complete (own ops) and visible (barrier).
  asm volatile("s_wait_asynccnt 0" ::: "memory");
  __syncthreads();

  // Epilogue: add gate-weighted b2 (from LDS), store f32 output.
  #pragma unroll
  for (int n = 0; n < 8; ++n) {
    const int d = dcol0 + n * 16 + lm;
    #pragma unroll
    for (int r = 0; r < 8; ++r) {
      const int m = r + 8 * lh;
      float bias = 0.f;
      #pragma unroll
      for (int e2 = 0; e2 < Ee; ++e2) bias += gs[m][e2] * b2s[e2 * Dd + d];
      out[(size_t)(token0 + m) * Dd + d] = acc[n][r] + bias;
    }
  }
}

// ---------------------------------------------------------------------------
// Launch
// ---------------------------------------------------------------------------
static inline size_t align256(size_t v) { return (v + 255) & ~(size_t)255; }

extern "C" void kernel_launch(void* const* d_in, const int* in_sizes, int n_in,
                              void* d_out, int out_size, void* d_ws, size_t ws_size,
                              hipStream_t stream) {
  const float* x  = (const float*)d_in[0];
  const float* Wg = (const float*)d_in[1];
  const float* bg = (const float*)d_in[2];
  const float* W1 = (const float*)d_in[3];
  const float* b1 = (const float*)d_in[4];
  const float* W2 = (const float*)d_in[5];
  const float* b2 = (const float*)d_in[6];
  float* out = (float*)d_out;

  const int Ntok = in_sizes[0] / Dd;   // B*T = 4096

  // Workspace layout
  char* ws = (char*)d_ws;
  size_t off = 0;
  float*  gbuf = (float*)(ws + off);  off += align256((size_t)Ntok * Ee * 4);
  __bf16* xbuf = (__bf16*)(ws + off); off += align256((size_t)Ntok * Dd * 2);
  __bf16* W1T  = (__bf16*)(ws + off); off += align256((size_t)Ee * Hh * Dd * 2);
  __bf16* W2T  = (__bf16*)(ws + off); off += align256((size_t)Ee * Dd * Hh * 2);

  // 1) x -> bf16
  cvt_x_kernel<<<1024, 256, 0, stream>>>(x, xbuf, Ntok * Dd);
  // 2) W1 [E,D,H] -> [E,H,D] bf16 ; W2 [E,H,D] -> [E,D,H] bf16
  dim3 tb(32, 8);
  transpose_cvt_kernel<<<dim3(Hh / 32, Dd / 32, Ee), tb, 0, stream>>>(W1, W1T, Dd, Hh);
  transpose_cvt_kernel<<<dim3(Dd / 32, Hh / 32, Ee), tb, 0, stream>>>(W2, W2T, Hh, Dd);
  // 3) gating (one wave per token)
  gate_kernel<<<(Ntok + 7) / 8, 256, 0, stream>>>(x, Wg, bg, gbuf, Ntok);
  // 4) fused dual-GEMM FFN + combine
  moe_ffn_kernel<<<Ntok / TM, 256, 0, stream>>>(xbuf, W1T, W2T, b1, b2, gbuf, out);
}
